// GCNEncoder_68865505624262
// MI455X (gfx1250) — matrix operand order repaired and checked by reading, not compiled, not run on hardware
//
#include <hip/hip_runtime.h>

#define NNODES 50000
#define FIN    96
#define HID    128
#define FOUT   64

typedef __attribute__((ext_vector_type(16))) __bf16 v16bf;
typedef __attribute__((ext_vector_type(8)))  __bf16 v8bf;
typedef __attribute__((ext_vector_type(8)))  float  v8f;

// ---------------- WMMA fragment loaders (CDNA5 16x16x32 bf16 layouts) ----------------

// A 16x32: lanes 0-15: M=lane, K in {0..7} (V0-3) and {16..23} (V4-7);
//          lanes16-31: M=lane-16, K in {8..15} and {24..31}.
// bf16 source: each K-octet is 16 contiguous bytes -> one b128 load.
__device__ __forceinline__ v16bf load_a_frag_bf(const __bf16* __restrict__ A, int lda) {
  const int lane  = threadIdx.x & 31;
  const int m     = lane & 15;
  const int khalf = (lane < 16) ? 0 : 8;
  const __bf16* row = A + (long long)m * lda + khalf;
  v8bf lo = *(const v8bf*)(row);        // K = khalf .. khalf+7
  v8bf hi = *(const v8bf*)(row + 16);   // K = 16+khalf .. 16+khalf+7
  v16bf a;
#pragma unroll
  for (int i = 0; i < 8; ++i) { a[i] = lo[i]; a[8 + i] = hi[i]; }
  return a;
}

// A fragment from f32 source: contiguous loads + native f32->bf16 converts.
__device__ __forceinline__ v16bf load_a_frag_f32(const float* __restrict__ A, int lda) {
  const int lane  = threadIdx.x & 31;
  const int m     = lane & 15;
  const int khalf = (lane < 16) ? 0 : 8;
  const float* row = A + (long long)m * lda + khalf;
  v16bf a;
#pragma unroll
  for (int i = 0; i < 8; ++i) {
    a[i]     = (__bf16)row[i];
    a[8 + i] = (__bf16)row[16 + i];
  }
  return a;
}

// B 32x16 (B[k][n] = W[n][k], W bf16 row-major [Nout,K]):
// lanes 0-15: N=lane, K in {0..7} and {16..23}; lanes 16-31: N=lane-16, K in {8..15} and {24..31}.
// two K-octets of 8 contiguous bf16 per lane -> two b128 loads.
__device__ __forceinline__ v16bf load_b_frag_bf(const __bf16* __restrict__ W, int ldw) {
  const int lane  = threadIdx.x & 31;
  const int n     = lane & 15;
  const int khalf = (lane < 16) ? 0 : 8;
  const __bf16* row = W + (long long)n * ldw + khalf;
  v8bf lo = *(const v8bf*)(row);
  v8bf hi = *(const v8bf*)(row + 16);
  v16bf b;
#pragma unroll
  for (int i = 0; i < 8; ++i) { b[i] = lo[i]; b[8 + i] = hi[i]; }
  return b;
}

// ---------------- Utility kernels ----------------

__global__ void zero_kernel(float* __restrict__ p, long long n) {
  long long i = (long long)blockIdx.x * blockDim.x + threadIdx.x;
  const long long stride = (long long)gridDim.x * blockDim.x;
  for (; i < n; i += stride) p[i] = 0.0f;
}

__global__ void f32_to_bf16_kernel(const float* __restrict__ src,
                                   __bf16* __restrict__ dst, int n) {
  int i = blockIdx.x * blockDim.x + threadIdx.x;
  if (i < n) dst[i] = (__bf16)src[i];
}

// ---------------- Edge scatter kernels (one wave per edge) ----------------

__global__ __launch_bounds__(256) void scatter1_kernel(
    const float* __restrict__ x, const int* __restrict__ src,
    const int* __restrict__ dst, float* __restrict__ agg, int E) {
  const int wave = (int)((blockIdx.x * blockDim.x + threadIdx.x) >> 5);
  const int lane = threadIdx.x & 31;
  if (wave >= E) return;
  const int s = src[wave], d = dst[wave];
  if (s == d) return;  // remove_self_loops
  const float* __restrict__ xs = x   + (long long)s * FIN;
  float* __restrict__       ad = agg + (long long)d * FIN;
#pragma unroll
  for (int i = 0; i < FIN / 32; ++i)
    atomicAdd(&ad[lane + 32 * i], xs[lane + 32 * i]);
}

__global__ __launch_bounds__(256) void scatter2_kernel(
    const float* __restrict__ p2, const int* __restrict__ src,
    const int* __restrict__ dst, float* __restrict__ out, int E) {
  const int wave = (int)((blockIdx.x * blockDim.x + threadIdx.x) >> 5);
  const int lane = threadIdx.x & 31;
  if (wave >= E) return;
  const int s = src[wave], d = dst[wave];
  if (s == d) return;
  const float* __restrict__ ps = p2  + (long long)s * FOUT;
  float* __restrict__       od = out + (long long)d * FOUT;
#pragma unroll
  for (int i = 0; i < FOUT / 32; ++i)
    atomicAdd(&od[lane + 32 * i], ps[lane + 32 * i]);
}

// ---------------- Layer 1: h = relu([agg1|x] @ [W1_rel|W1_root]^T + b1) ----------------
// grid.x = NNODES/16; 8 waves/block, wave w owns output cols [16w,16w+16).
// Output h stored directly in bf16 (consumed only as layer-2 A operand).
__global__ __launch_bounds__(256) void gemm1_kernel(
    const float* __restrict__ agg1, const float* __restrict__ x,
    const __bf16* __restrict__ W1_rel_bf, const __bf16* __restrict__ W1_root_bf,
    const float* __restrict__ b1, __bf16* __restrict__ h) {
  const int node0 = blockIdx.x * 16;
  const int wave  = threadIdx.x >> 5;
  const int lane  = threadIdx.x & 31;
  const int jbase = wave * 16;

  v8f c = {};
#pragma unroll
  for (int kt = 0; kt < FIN; kt += 32) {
    v16bf a = load_a_frag_f32(agg1 + (long long)node0 * FIN + kt, FIN);
    v16bf b = load_b_frag_bf(W1_rel_bf + (long long)jbase * FIN + kt, FIN);
    c = __builtin_amdgcn_wmma_f32_16x16x32_bf16(false, a, false, b, (short)0, c, false, false);
  }
#pragma unroll
  for (int kt = 0; kt < FIN; kt += 32) {
    v16bf a = load_a_frag_f32(x + (long long)node0 * FIN + kt, FIN);
    v16bf b = load_b_frag_bf(W1_root_bf + (long long)jbase * FIN + kt, FIN);
    c = __builtin_amdgcn_wmma_f32_16x16x32_bf16(false, a, false, b, (short)0, c, false, false);
  }

  // C/D layout: N = lane&15; row r -> M = r + 8*(lane>=16)
  const int   n    = lane & 15;
  const int   mofs = (lane < 16) ? 0 : 8;
  const float bias = b1[jbase + n];
#pragma unroll
  for (int r = 0; r < 8; ++r) {
    float v = c[r] + bias;
    v = v > 0.0f ? v : 0.0f;
    h[(long long)(node0 + r + mofs) * HID + jbase + n] = (__bf16)v;
  }
}

// ---------------- Layer 2: p2 = h@W2_rel^T ; out = h@W2_root^T + b2 ----------------
// waves 0-3: relational projection (scattered later); waves 4-7: root term into d_out.
__global__ __launch_bounds__(256) void gemm2_kernel(
    const __bf16* __restrict__ h,
    const __bf16* __restrict__ W2_rel_bf, const __bf16* __restrict__ W2_root_bf,
    const float* __restrict__ b2, float* __restrict__ p2, float* __restrict__ out) {
  const int node0   = blockIdx.x * 16;
  const int wave    = threadIdx.x >> 5;
  const int lane    = threadIdx.x & 31;
  const bool isroot = wave >= 4;
  const int jbase   = (isroot ? wave - 4 : wave) * 16;
  const __bf16* __restrict__ W = isroot ? W2_root_bf : W2_rel_bf;

  v8f c = {};
#pragma unroll
  for (int kt = 0; kt < HID; kt += 32) {
    v16bf a = load_a_frag_bf(h + (long long)node0 * HID + kt, HID);
    v16bf b = load_b_frag_bf(W + (long long)jbase * HID + kt, HID);
    c = __builtin_amdgcn_wmma_f32_16x16x32_bf16(false, a, false, b, (short)0, c, false, false);
  }

  const int n    = lane & 15;
  const int mofs = (lane < 16) ? 0 : 8;
  if (isroot) {
    const float bias = b2[jbase + n];
#pragma unroll
    for (int r = 0; r < 8; ++r)
      out[(long long)(node0 + r + mofs) * FOUT + jbase + n] = c[r] + bias;
  } else {
#pragma unroll
    for (int r = 0; r < 8; ++r)
      p2[(long long)(node0 + r + mofs) * FOUT + jbase + n] = c[r];
  }
}

// ---------------- Host launcher ----------------

extern "C" void kernel_launch(void* const* d_in, const int* in_sizes, int n_in,
                              void* d_out, int out_size, void* d_ws, size_t ws_size,
                              hipStream_t stream) {
  const float* x       = (const float*)d_in[0];
  const int*   edge    = (const int*)d_in[1];
  const float* W1_rel  = (const float*)d_in[2];
  const float* b1      = (const float*)d_in[3];
  const float* W1_root = (const float*)d_in[4];
  const float* W2_rel  = (const float*)d_in[5];
  const float* b2      = (const float*)d_in[6];
  const float* W2_root = (const float*)d_in[7];
  float* out = (float*)d_out;

  const int E = in_sizes[1] / 2;
  const int* src = edge;
  const int* dst = edge + E;

  // Workspace layout (all segments 16B-aligned):
  //   agg1 [N,96]  f32   19.2 MB
  //   h    [N,128] bf16  12.8 MB
  //   p2   [N,64]  f32   12.8 MB
  //   bf16 weight copies: W1_rel, W1_root [128,96]; W2_rel, W2_root [64,128]
  char* wsp = (char*)d_ws;
  float*  agg1      = (float*)wsp;                wsp += (size_t)NNODES * FIN  * sizeof(float);
  __bf16* h         = (__bf16*)wsp;               wsp += (size_t)NNODES * HID  * sizeof(__bf16);
  float*  p2        = (float*)wsp;                wsp += (size_t)NNODES * FOUT * sizeof(float);
  __bf16* W1rel_bf  = (__bf16*)wsp;               wsp += (size_t)HID * FIN  * sizeof(__bf16);
  __bf16* W1root_bf = (__bf16*)wsp;               wsp += (size_t)HID * FIN  * sizeof(__bf16);
  __bf16* W2rel_bf  = (__bf16*)wsp;               wsp += (size_t)FOUT * HID * sizeof(__bf16);
  __bf16* W2root_bf = (__bf16*)wsp;

  // 0) stage weights in bf16 (tiny; removes all conversion work from GEMM loops)
  f32_to_bf16_kernel<<<(HID * FIN  + 255) / 256, 256, 0, stream>>>(W1_rel,  W1rel_bf,  HID * FIN);
  f32_to_bf16_kernel<<<(HID * FIN  + 255) / 256, 256, 0, stream>>>(W1_root, W1root_bf, HID * FIN);
  f32_to_bf16_kernel<<<(FOUT * HID + 255) / 256, 256, 0, stream>>>(W2_rel,  W2rel_bf,  FOUT * HID);
  f32_to_bf16_kernel<<<(FOUT * HID + 255) / 256, 256, 0, stream>>>(W2_root, W2root_bf, FOUT * HID);

  // 1) zero layer-1 accumulator (every launch; atomics accumulate into it)
  zero_kernel<<<2048, 256, 0, stream>>>(agg1, (long long)NNODES * FIN);

  // 2) layer-1 edge aggregation (96-wide f32 atomics; working set lives in L2)
  const int eblocks = (E + 7) / 8;  // 8 waves per 256-thread block
  scatter1_kernel<<<eblocks, 256, 0, stream>>>(x, src, dst, agg1, E);

  // 3) h = relu([agg1|x] @ [W1_rel|W1_root]^T + b1)  -> bf16
  gemm1_kernel<<<NNODES / 16, 256, 0, stream>>>(agg1, x, W1rel_bf, W1root_bf, b1, h);

  // 4) project BEFORE aggregating: p2 = h@W2_rel^T ; out = h@W2_root^T + b2
  gemm2_kernel<<<NNODES / 16, 256, 0, stream>>>(h, W2rel_bf, W2root_bf, b2, p2, out);

  // 5) layer-2 edge aggregation of 64-wide projected messages into out
  scatter2_kernel<<<eblocks, 256, 0, stream>>>(p2, src, dst, out, E);
}
